// HypernetworkGatedDANN_16277926052500
// MI455X (gfx1250) — compile-verified, matrix operand order checked by reading, not run on hardware
//
#include <hip/hip_runtime.h>
#include <hip/hip_bf16.h>

// ---------------------------------------------------------------------------
// Problem constants (from reference)
// ---------------------------------------------------------------------------
#define BATCH 4096
#define INDIM 256
#define CTXD  64
#define OUTD  128
#define NN    128
#define KK    4
#define JJ    64
#define NG    512            // N*K
#define WSZ   131072         // IN*NG  (Wg elements per layer)
#define PER   131584         // WSZ + NG
#define TPTOT 263168         // 2*PER

typedef unsigned short ushort_t;
typedef __attribute__((ext_vector_type(16))) __bf16 v16bf;
typedef __attribute__((ext_vector_type(8)))  float  v8f;
typedef __attribute__((ext_vector_type(4)))  unsigned int u32x4;

union FragU { v16bf v; u32x4 q[2]; };

// f32 -> bf16 round-to-nearest-even
__device__ __forceinline__ ushort_t f2bf(float f) {
    unsigned int u = __float_as_uint(f);
    u += 0x7FFFu + ((u >> 16) & 1u);
    return (ushort_t)(u >> 16);
}

// Load one 16x32 bf16 WMMA operand fragment (A-layout; B uses the symmetric
// per-column layout which has the identical per-lane addressing pattern since
// our B sources are stored transposed = rows contiguous along the GEMM-K dim).
// lane 0-15 : row/col = lane,     K = ks..ks+7  and ks+16..ks+23
// lane16-31 : row/col = lane-16,  K = ks+8..ks+15 and ks+24..ks+31  (half=1 -> ks+=8)
__device__ __forceinline__ v16bf load_frag(const ushort_t* base, int ld, int rc, int ks) {
    const ushort_t* p = base + (size_t)rc * ld + ks;
    FragU u;
    u.q[0] = *(const u32x4*)(p);
    u.q[1] = *(const u32x4*)(p + 16);
    return u.v;
}

// ---------------------------------------------------------------------------
// Hypernet stage 1+2: context -> hv2[64]
// ---------------------------------------------------------------------------
__global__ void k_hyper(const float* __restrict__ ctx,
                        const float* __restrict__ h1w, const float* __restrict__ h1b,
                        const float* __restrict__ h2w, const float* __restrict__ h2b,
                        float* __restrict__ hv2) {
    __shared__ float s1[32];
    int tid = threadIdx.x;
    if (tid < 32) {
        float acc = 0.f;
        #pragma unroll 8
        for (int i = 0; i < CTXD; ++i) acc += ctx[i] * h1w[tid * CTXD + i];
        s1[tid] = fmaxf(acc + h1b[tid], 0.f);
    }
    __syncthreads();
    if (tid < 64) {
        float acc = 0.f;
        #pragma unroll 8
        for (int i = 0; i < 32; ++i) acc += s1[i] * h2w[tid * 32 + i];
        hv2[tid] = fmaxf(acc + h2b[tid], 0.f);
    }
}

// ---------------------------------------------------------------------------
// Hypernet stage 3: gw = hv2 @ h3w.T + h3b  (bandwidth bound: 67MB of h3w)
// Also emits bf16 copies of the two dynamic gate-weight matrices.
// ---------------------------------------------------------------------------
__global__ void k_gw(const float* __restrict__ hv2,
                     const float* __restrict__ h3w, const float* __restrict__ h3b,
                     float* __restrict__ gw,
                     ushort_t* __restrict__ wg0, ushort_t* __restrict__ wg1) {
    __shared__ float s[64];
    int tid = threadIdx.x;
    if (tid < 64) s[tid] = hv2[tid];
    __syncthreads();
    size_t idx = (size_t)blockIdx.x * 256 + tid;       // grid sized exactly TP/256
    const float* row = h3w + idx * 64;
    __builtin_prefetch(row + 64, 0, 1);                // global_prefetch_b8
    float acc = 0.f;
    #pragma unroll
    for (int i = 0; i < 64; ++i) acc += row[i] * s[i];
    float v = acc + h3b[idx];
    gw[idx] = v;
    if (idx < WSZ)                                  wg0[idx]       = f2bf(v);
    else if (idx >= PER && idx < PER + (size_t)WSZ) wg1[idx - PER] = f2bf(v);
}

// ---------------------------------------------------------------------------
// Elementwise f32 -> bf16 conversion
// ---------------------------------------------------------------------------
__global__ void k_cvt(const float* __restrict__ src, ushort_t* __restrict__ dst, int n) {
    int i = blockIdx.x * blockDim.x + threadIdx.x;
    int stride = gridDim.x * blockDim.x;
    for (; i < n; i += stride) dst[i] = f2bf(src[i]);
}

// ---------------------------------------------------------------------------
// Gates: G[b, ng] = sigmoid( X(4096x256) @ Wg(512x256)^T + bg )
// One wave per 16x16 output tile, 8 bf16 WMMA k-steps.
// ---------------------------------------------------------------------------
__global__ void k_gates(const ushort_t* __restrict__ xbf,
                        const ushort_t* __restrict__ wg,
                        const float* __restrict__ bg,
                        float* __restrict__ gates) {
    int lane = threadIdx.x & 31;
    int wid  = threadIdx.x >> 5;
    int half = lane >> 4, lq = lane & 15;
    int bm = (blockIdx.x * 4 + wid) * 16;
    int bn = blockIdx.y * 16;

    v16bf a[8];
    #pragma unroll
    for (int kc = 0; kc < 8; ++kc)
        a[kc] = load_frag(xbf, INDIM, bm + lq, kc * 32 + half * 8);

    v8f c = {0.f,0.f,0.f,0.f,0.f,0.f,0.f,0.f};
    #pragma unroll
    for (int kc = 0; kc < 8; ++kc) {
        v16bf b = load_frag(wg, INDIM, bn + lq, kc * 32 + half * 8);
        c = __builtin_amdgcn_wmma_f32_16x16x32_bf16(false, a[kc], false, b,
                                                    (short)0, c, false, false);
    }
    int col  = bn + lq;
    float bv = bg[col];
    #pragma unroll
    for (int r = 0; r < 8; ++r) {
        int row = bm + r + 8 * half;
        float v = c[r] + bv;
        gates[(size_t)row * NG + col] = 1.0f / (1.0f + __expf(-v));
    }
}

// ---------------------------------------------------------------------------
// Fused DANN layer core:
//   out[b,n] = sum_kj relu( cur @ Wb[n]^T + bb[n] )[kj] * gate[b,n,kj/64] * Ws[n,kj] + bs[n]
// One wave handles a 32-row batch tile (two 16-row WMMA M-tiles) for one n:
// each Wb[n] B-fragment is loaded once and feeds two back-to-back WMMAs
// (halves L2 traffic on the dominant, non-reused operand). Epilogue
// (bias -> relu -> xgate -> xWs) is fused in registers; a width-16 butterfly
// reduction folds the 16 kj-columns held across lanes.
// ---------------------------------------------------------------------------
template <int INCH>   // IN/32 : 8 for layer0 (IN=256), 4 for layer1 (IN=128)
__global__ void k_layer(const ushort_t* __restrict__ curA,
                        const ushort_t* __restrict__ wb,
                        const float* __restrict__ bb,
                        const float* __restrict__ ws,
                        const float* __restrict__ bs,
                        const float* __restrict__ gates,
                        ushort_t* __restrict__ outbf) {
    const int INE = INCH * 32;
    int lane = threadIdx.x & 31;
    int wid  = threadIdx.x >> 5;
    int half = lane >> 4, lq = lane & 15;
    int bm = blockIdx.x * 128 + wid * 32;     // 32 rows per wave, 128 per block
    int n  = blockIdx.y;
    const ushort_t* wbn = wb + (size_t)n * 256 * INE;
    const float* bbn = bb + n * 256;
    const float* wsn = ws + n * 256;

    v16bf a[2][INCH];
    #pragma unroll
    for (int mt = 0; mt < 2; ++mt)
        #pragma unroll
        for (int kc = 0; kc < INCH; ++kc)
            a[mt][kc] = load_frag(curA, INE, bm + mt * 16 + lq, kc * 32 + half * 8);

    float acc[2][8] = {};
    float gk[2][8];

    #pragma unroll
    for (int t = 0; t < 16; ++t) {
        if ((t & 3) == 0) {                  // gate index k advances every 4 tiles (J=64)
            int k = t >> 2;
            #pragma unroll
            for (int mt = 0; mt < 2; ++mt)
                #pragma unroll
                for (int r = 0; r < 8; ++r)
                    gk[mt][r] = gates[(size_t)(bm + mt * 16 + r + 8 * half) * NG + n * KK + k];
        }
        int   kj  = t * 16 + lq;
        float bbv = bbn[kj];
        float wsv = wsn[kj];

        v8f c0 = {0.f,0.f,0.f,0.f,0.f,0.f,0.f,0.f};
        v8f c1 = {0.f,0.f,0.f,0.f,0.f,0.f,0.f,0.f};
        #pragma unroll
        for (int kc = 0; kc < INCH; ++kc) {
            v16bf b = load_frag(wbn, INE, kj, kc * 32 + half * 8);
            c0 = __builtin_amdgcn_wmma_f32_16x16x32_bf16(false, a[0][kc], false, b,
                                                         (short)0, c0, false, false);
            c1 = __builtin_amdgcn_wmma_f32_16x16x32_bf16(false, a[1][kc], false, b,
                                                         (short)0, c1, false, false);
        }
        #pragma unroll
        for (int r = 0; r < 8; ++r) {
            float v0 = fmaxf(c0[r] + bbv, 0.f);
            float v1 = fmaxf(c1[r] + bbv, 0.f);
            acc[0][r] += v0 * gk[0][r] * wsv;
            acc[1][r] += v1 * gk[1][r] * wsv;
        }
    }

    // fold the 16 kj-columns (one per lane within each half) per row
    float bsv = bs[n];
    #pragma unroll
    for (int mt = 0; mt < 2; ++mt) {
        #pragma unroll
        for (int r = 0; r < 8; ++r) {
            float v = acc[mt][r];
            v += __shfl_xor(v, 8, 16);
            v += __shfl_xor(v, 4, 16);
            v += __shfl_xor(v, 2, 16);
            v += __shfl_xor(v, 1, 16);
            acc[mt][r] = v;
        }
        if (lq == 0) {
            #pragma unroll
            for (int r = 0; r < 8; ++r) {
                int row = bm + mt * 16 + r + 8 * half;
                outbf[(size_t)row * NN + n] = f2bf(acc[mt][r] + bsv);
            }
        }
    }
}

// ---------------------------------------------------------------------------
// Output GEMM: out = cur2(4096x128) @ Wout(128x128)^T + bout  (f32 out)
// ---------------------------------------------------------------------------
__global__ void k_final(const ushort_t* __restrict__ cur,
                        const ushort_t* __restrict__ wout,
                        const float* __restrict__ bout,
                        float* __restrict__ out) {
    int lane = threadIdx.x & 31;
    int wid  = threadIdx.x >> 5;
    int half = lane >> 4, lq = lane & 15;
    int bm = (blockIdx.x * 4 + wid) * 16;
    int bo = blockIdx.y * 16;

    v16bf a[4];
    #pragma unroll
    for (int kc = 0; kc < 4; ++kc)
        a[kc] = load_frag(cur, NN, bm + lq, kc * 32 + half * 8);

    v8f c = {0.f,0.f,0.f,0.f,0.f,0.f,0.f,0.f};
    #pragma unroll
    for (int kc = 0; kc < 4; ++kc) {
        v16bf b = load_frag(wout, NN, bo + lq, kc * 32 + half * 8);
        c = __builtin_amdgcn_wmma_f32_16x16x32_bf16(false, a[kc], false, b,
                                                    (short)0, c, false, false);
    }
    int col  = bo + lq;
    float bv = bout[col];
    #pragma unroll
    for (int r = 0; r < 8; ++r) {
        int row = bm + r + 8 * half;
        out[(size_t)row * OUTD + col] = c[r] + bv;
    }
}

// ---------------------------------------------------------------------------
// Host launcher
// ---------------------------------------------------------------------------
static constexpr size_t align256(size_t x) { return (x + 255) & ~(size_t)255; }

extern "C" void kernel_launch(void* const* d_in, const int* in_sizes, int n_in,
                              void* d_out, int out_size, void* d_ws, size_t ws_size,
                              hipStream_t stream) {
    const float* x    = (const float*)d_in[0];
    const float* ctx  = (const float*)d_in[1];
    const float* Wb0  = (const float*)d_in[2];
    const float* bb0  = (const float*)d_in[3];
    const float* Ws0  = (const float*)d_in[4];
    const float* bs0  = (const float*)d_in[5];
    const float* Wb1  = (const float*)d_in[6];
    const float* bb1  = (const float*)d_in[7];
    const float* Ws1  = (const float*)d_in[8];
    const float* bs1  = (const float*)d_in[9];
    const float* Wout = (const float*)d_in[10];
    const float* bout = (const float*)d_in[11];
    const float* h1w  = (const float*)d_in[12];
    const float* h1b  = (const float*)d_in[13];
    const float* h2w  = (const float*)d_in[14];
    const float* h2b  = (const float*)d_in[15];
    const float* h3w  = (const float*)d_in[16];
    const float* h3b  = (const float*)d_in[17];
    float* out = (float*)d_out;

    // workspace carve-up (all 256B aligned; bf16 rows stay 16B aligned)
    char* base = (char*)d_ws;
    size_t off = 0;
    float*    hv2   = (float*)(base + off);    off = align256(off + 64 * 4);
    float*    gw    = (float*)(base + off);    off = align256(off + (size_t)TPTOT * 4);
    ushort_t* wg0   = (ushort_t*)(base + off); off = align256(off + (size_t)WSZ * 2);
    ushort_t* wg1   = (ushort_t*)(base + off); off = align256(off + (size_t)WSZ * 2);
    ushort_t* xbf   = (ushort_t*)(base + off); off = align256(off + (size_t)BATCH * INDIM * 2);
    ushort_t* wb0b  = (ushort_t*)(base + off); off = align256(off + (size_t)NN * 256 * INDIM * 2);
    ushort_t* wb1b  = (ushort_t*)(base + off); off = align256(off + (size_t)NN * 256 * NN * 2);
    ushort_t* woutb = (ushort_t*)(base + off); off = align256(off + (size_t)OUTD * NN * 2);
    float*    g0    = (float*)(base + off);    off = align256(off + (size_t)BATCH * NG * 4);
    float*    g1    = (float*)(base + off);    off = align256(off + (size_t)BATCH * NG * 4);
    ushort_t* c1bf  = (ushort_t*)(base + off); off = align256(off + (size_t)BATCH * NN * 2);
    ushort_t* c2bf  = (ushort_t*)(base + off); off = align256(off + (size_t)BATCH * NN * 2);
    (void)ws_size; (void)n_in; (void)in_sizes; (void)out_size;

    // 1. hypernet MLP -> hv2
    k_hyper<<<1, 64, 0, stream>>>(ctx, h1w, h1b, h2w, h2b, hv2);

    // 2. gw = hv2 @ h3w.T + h3b (+ bf16 gate-weight copies)
    k_gw<<<TPTOT / 256, 256, 0, stream>>>(hv2, h3w, h3b, gw, wg0, wg1);

    // 3. one-time bf16 conversions of static operands
    k_cvt<<<4096, 256, 0, stream>>>(x,    xbf,   BATCH * INDIM);
    k_cvt<<<8192, 256, 0, stream>>>(Wb0,  wb0b,  NN * 256 * INDIM);
    k_cvt<<<8192, 256, 0, stream>>>(Wb1,  wb1b,  NN * 256 * NN);
    k_cvt<<<64,   256, 0, stream>>>(Wout, woutb, OUTD * NN);

    // 4. gate GEMMs (both layers gate from x)
    k_gates<<<dim3(BATCH / 64, NG / 16), 128, 0, stream>>>(xbf, wg0, gw + WSZ,       g0);
    k_gates<<<dim3(BATCH / 64, NG / 16), 128, 0, stream>>>(xbf, wg1, gw + PER + WSZ, g1);

    // 5. fused DANN layers (32 rows per wave, B-fragment reuse x2)
    k_layer<8><<<dim3(BATCH / 128, NN), 128, 0, stream>>>(xbf,  wb0b, bb0, Ws0, bs0, g0, c1bf);
    k_layer<4><<<dim3(BATCH / 128, NN), 128, 0, stream>>>(c1bf, wb1b, bb1, Ws1, bs1, g1, c2bf);

    // 6. output projection
    k_final<<<dim3(BATCH / 64, OUTD / 16), 128, 0, stream>>>(c2bf, woutb, bout, out);
}
